// mLSTMblock_86844238725688
// MI455X (gfx1250) — matrix operand order repaired and checked by reading, not compiled, not run on hardware
//
#include <hip/hip_runtime.h>
#include <hip/hip_bf16.h>

// mLSTM block for MI455X (gfx1250, wave32, WMMA).
// - V_w cast once to bf16 (134MB, fits 192MB L2) -> per-step streaming of V hits L2.
// - All GEMMs via v_wmma_f32_16x16x32_bf16, fp32 accumulation.
// - The (16 x 131072)@(131072 x 512) per-step GEMM synthesizes its A-matrix
//   (outer product x_t x h) in registers with packed bf16 muls, reused across
//   4 N-tiles per wave; split-K=128 partials + small fused update kernel.
// - T=512 steps serialized by stream order (graph-capture safe).

#define B_  16
#define T_  512
#define D_  256
#define H_  512
#define G_  2048            // 4*H
#define DH_ 131072ULL       // D*H
#define KSPLIT 128          // K-slices for the V GEMM (each slice spans 1024 k)
#define NTW 4               // N-tiles per mv wave (A-fragment reuse)

typedef __attribute__((ext_vector_type(16))) __bf16 v16bf;
typedef __attribute__((ext_vector_type(8)))  float  v8f;

static __device__ __forceinline__ v16bf ld_pair(const __bf16* p0, const __bf16* p1) {
  union { uint4 u[2]; v16bf v; } r;
  r.u[0] = *(const uint4*)p0;   // 16B aligned by construction
  r.u[1] = *(const uint4*)p1;
  return r.v;
}

// A fragment (16x32, bf16) from row-major [rows x lda].
// lane<16 -> M=lane, K = j (j<8) and 16+j; lane>=16 -> M=lane-16, K = 8+j and 24+j.
static __device__ __forceinline__ v16bf load_a(const __bf16* A, int lda, int m0, int k0, int lane) {
  int m  = m0 + (lane & 15);
  int kb = k0 + ((lane >> 4) << 3);
  const __bf16* p = A + (size_t)m * lda + kb;
  return ld_pair(p, p + 16);
}

// B fragment (32x16) where B[k][n] = S[n][k], S row-major [N x lds] (n-major weights).
// lane<16 -> K = j (j<8) and 16+j; lane>=16 -> K = 8+j and 24+j; N = lane&15.
static __device__ __forceinline__ v16bf load_b(const __bf16* S, size_t lds, int n0, int k0, int lane) {
  int n  = n0 + (lane & 15);
  int kb = k0 + ((lane >> 4) << 3);
  const __bf16* p = S + (size_t)n * lds + kb;
  return ld_pair(p, p + 16);
}

// C/D f32 16x16: VGPR r: lane<16 -> M=r, N=lane; lane>=16 -> M=8+r, N=lane-16.
static __device__ __forceinline__ void store_c(float* C, int ldc, int m0, int n0, v8f c, int lane) {
  int n  = n0 + (lane & 15);
  int mb = m0 + ((lane >> 4) << 3);
  #pragma unroll
  for (int r = 0; r < 8; ++r) C[(size_t)(mb + r) * ldc + n] = c[r];
}

static __device__ __forceinline__ v16bf splat16(__bf16 x) {
  v16bf v;
  #pragma unroll
  for (int j = 0; j < 16; ++j) v[j] = x;
  return v;
}

// ---------------- prologue kernels ----------------

__global__ void f32_to_bf16_k(const float* __restrict__ s, __bf16* __restrict__ d, size_t n) {
  size_t i  = (size_t)blockIdx.x * blockDim.x + threadIdx.x;
  size_t st = (size_t)gridDim.x * blockDim.x;
  for (; i < n; i += st) d[i] = (__bf16)s[i];
}

__global__ void init_state_k(float* __restrict__ h, float* __restrict__ c, __bf16* __restrict__ hbf) {
  int i = blockIdx.x * blockDim.x + threadIdx.x;
  if (i < B_ * H_) { h[i] = 0.f; c[i] = 0.f; hbf[i] = (__bf16)0.f; }
}

// wx_all = x @ W_w.T + W_b : (8192 x 2048), K=256. One 16x16 tile per wave.
__global__ void wx_gemm_k(const __bf16* __restrict__ xbf, const __bf16* __restrict__ Wbf,
                          const float* __restrict__ Wb, float* __restrict__ wx) {
  int lane = threadIdx.x & 31;
  int w    = blockIdx.x * (blockDim.x >> 5) + (threadIdx.x >> 5);
  int mt   = w >> 7;    // 512 row tiles (rows = b*T+t)
  int nt   = w & 127;   // 128 col tiles
  v8f acc = {};
  #pragma unroll
  for (int k0 = 0; k0 < D_; k0 += 32) {
    v16bf a = load_a(xbf, D_, mt * 16, k0, lane);
    v16bf b = load_b(Wbf, (size_t)D_, nt * 16, k0, lane);
    acc = __builtin_amdgcn_wmma_f32_16x16x32_bf16(false, a, false, b, (short)0, acc, false, false);
  }
  float bias = Wb[nt * 16 + (lane & 15)];
  #pragma unroll
  for (int r = 0; r < 8; ++r) acc[r] += bias;
  store_c(wx, G_, mt * 16, nt * 16, acc, lane);
}

// ---------------- per-step kernels ----------------

// Blocks 0..15   : gates = wx[:,t,:] + h @ U_w.T + U_b  (16x2048, K=512) -> 128 tile-waves
// Blocks 16..143 : split-K V GEMM, A synthesized as x[b,d]*h[b,hh] with packed bf16,
//                  reused across NTW=4 N-tiles -> 8 ntile-groups x 128 kslices = 1024 waves
__global__ void step_compute_k(int t,
    const __bf16* __restrict__ xbf, const __bf16* __restrict__ hbf,
    const __bf16* __restrict__ Ubf, const float* __restrict__ Ub,
    const float* __restrict__ wx,   const __bf16* __restrict__ Vbf,
    float* __restrict__ gates, float* __restrict__ partials) {
  int lane = threadIdx.x & 31;
  int widx = threadIdx.x >> 5;

  if (blockIdx.x < 16) {
    int nt = blockIdx.x * 8 + widx;          // 0..127
    v8f acc = {};
    #pragma unroll 4
    for (int k0 = 0; k0 < H_; k0 += 32) {
      v16bf a = load_a(hbf, H_, 0, k0, lane);
      v16bf b = load_b(Ubf, (size_t)H_, nt * 16, k0, lane);
      acc = __builtin_amdgcn_wmma_f32_16x16x32_bf16(false, a, false, b, (short)0, acc, false, false);
    }
    int n  = nt * 16 + (lane & 15);
    int mb = (lane >> 4) << 3;
    float ub = Ub[n];
    #pragma unroll
    for (int r = 0; r < 8; ++r) {
      int m = mb + r;                        // batch index
      gates[(size_t)m * G_ + n] = acc[r] + ub + wx[((size_t)m * T_ + t) * G_ + n];
    }
  } else {
    int w   = (blockIdx.x - 16) * 8 + widx;  // 0..1023
    int ntg = w >> 7;                        // 0..7  (group of NTW=4 N-tiles)
    int ks  = w & (KSPLIT - 1);              // 0..127 k-slice
    int nt0 = ntg * NTW;
    int kbase = ks << 10;                    // * 1024 k per slice
    int b  = lane & 15;
    const __bf16* xrow = xbf + ((size_t)b * T_ + t) * D_;

    v8f acc[NTW];
    #pragma unroll
    for (int q = 0; q < NTW; ++q) acc[q] = (v8f){};

    #pragma unroll 4
    for (int kk = 0; kk < 1024; kk += 32) {
      int k0 = kbase + kk;
      int d  = k0 >> 9;                      // chunk of 32 never crosses d boundary (512%32==0)
      // A fragment of mult = x[b,d] * h[b, hh-window], as packed bf16 muls.
      v16bf hf = load_a(hbf, H_, 0, k0 & 511, lane);
      v16bf a  = splat16(xrow[d]) * hf;      // lowers to v_pk_mul_bf16 x8
      #pragma unroll
      for (int q = 0; q < NTW; ++q) {
        v16bf bb = load_b(Vbf, DH_, (nt0 + q) * 16, k0, lane);
        acc[q] = __builtin_amdgcn_wmma_f32_16x16x32_bf16(false, a, false, bb, (short)0, acc[q], false, false);
      }
    }
    #pragma unroll
    for (int q = 0; q < NTW; ++q)
      store_c(partials + ((size_t)((nt0 + q) * KSPLIT + ks) << 8), 16, 0, 0, acc[q], lane);
  }
}

// Reduce split-K partials, apply gates, update c/h, emit outs (and hT/cT tail).
__global__ void step_update_k(int t, const float* __restrict__ gates,
    const float* __restrict__ partials, const float* __restrict__ Vb,
    float* __restrict__ hbuf, float* __restrict__ cbuf, __bf16* __restrict__ hbf,
    float* __restrict__ out) {
  int idx = blockIdx.x * blockDim.x + threadIdx.x;  // 0..8191
  int b = idx >> 9, n = idx & 511;
  int nt = n >> 4, col = n & 15;

  float acc = Vb[n];
  const float* p = partials + ((size_t)nt * KSPLIT << 8) + b * 16 + col;
  #pragma unroll 8
  for (int ks = 0; ks < KSPLIT; ++ks) acc += p[(size_t)ks << 8];
  float m = tanhf(acc);

  float gi = gates[(size_t)b * G_ + n];
  float gf = gates[(size_t)b * G_ + 512 + n];
  float go = gates[(size_t)b * G_ + 1024 + n];
  float gg = gates[(size_t)b * G_ + 1536 + n];
  gi = 1.f / (1.f + __expf(-gi));
  gf = 1.f / (1.f + __expf(-gf));
  go = 1.f / (1.f + __expf(-go));
  gg = tanhf(gg);

  float c = gf * cbuf[idx] + gi * gg + 0.1f * m;
  float h = go * tanhf(c);
  cbuf[idx] = c;
  hbuf[idx] = h;
  hbf[idx]  = (__bf16)h;
  out[((size_t)b * T_ + t) * H_ + n] = h;

  const size_t BTH = (size_t)B_ * T_ * H_;
  out[BTH + idx] = h;                       // hT (last step's write survives)
  out[BTH + (size_t)B_ * H_ + idx] = c;     // cT
}

// ---------------- host launcher ----------------

extern "C" void kernel_launch(void* const* d_in, const int* in_sizes, int n_in,
                              void* d_out, int out_size, void* d_ws, size_t ws_size,
                              hipStream_t stream) {
  const float* x   = (const float*)d_in[0];
  const float* W_w = (const float*)d_in[1];
  const float* W_b = (const float*)d_in[2];
  const float* U_w = (const float*)d_in[3];
  const float* U_b = (const float*)d_in[4];
  const float* V_w = (const float*)d_in[5];
  const float* V_b = (const float*)d_in[6];
  float* out = (float*)d_out;

  char* ws = (char*)d_ws;
  size_t off = 0;
  auto carve = [&](size_t bytes) -> void* {
    void* p = ws + off;
    off = (off + bytes + 255) & ~(size_t)255;
    return p;
  };
  __bf16* Vbf   = (__bf16*)carve(DH_ * H_ * 2);            // 134 MB (L2-resident)
  __bf16* xbf   = (__bf16*)carve((size_t)B_ * T_ * D_ * 2);
  __bf16* Wbf   = (__bf16*)carve((size_t)G_ * D_ * 2);
  __bf16* Ubf   = (__bf16*)carve((size_t)G_ * H_ * 2);
  float*  wx    = (float*) carve((size_t)B_ * T_ * G_ * 4); // 67 MB
  float*  gates = (float*) carve((size_t)B_ * G_ * 4);
  float*  parts = (float*) carve((size_t)32 * KSPLIT * 256 * 4); // 4 MB
  float*  hbuf  = (float*) carve((size_t)B_ * H_ * 4);
  float*  cbuf  = (float*) carve((size_t)B_ * H_ * 4);
  __bf16* hbf   = (__bf16*)carve((size_t)B_ * H_ * 2);

  // Prologue: bf16 casts + zero state + big wx GEMM.
  f32_to_bf16_k<<<4096, 256, 0, stream>>>(V_w, Vbf, DH_ * H_);
  f32_to_bf16_k<<<1024, 256, 0, stream>>>(x,   xbf, (size_t)B_ * T_ * D_);
  f32_to_bf16_k<<<512,  256, 0, stream>>>(W_w, Wbf, (size_t)G_ * D_);
  f32_to_bf16_k<<<512,  256, 0, stream>>>(U_w, Ubf, (size_t)G_ * H_);
  init_state_k<<<(B_ * H_ + 255) / 256, 256, 0, stream>>>(hbuf, cbuf, hbf);
  wx_gemm_k<<<(512 * 128) / 8, 256, 0, stream>>>(xbf, Wbf, W_b, wx);

  // Recurrence: stream order provides the step-to-step dependency.
  const int mv_blocks = (8 * KSPLIT) / 8;   // 1024 waves / 8 per block = 128
  for (int t = 0; t < T_; ++t) {
    step_compute_k<<<16 + mv_blocks, 256, 0, stream>>>(
        t, xbf, hbf, Ubf, U_b, wx, Vbf, gates, parts);
    step_update_k<<<(B_ * H_) / 256, 256, 0, stream>>>(
        t, gates, parts, V_b, hbuf, cbuf, hbf, out);
  }
  (void)in_sizes; (void)n_in; (void)out_size; (void)ws_size;
}
